// TransformerDecoder_26852135534879
// MI455X (gfx1250) — matrix operand order repaired and checked by reading, not compiled, not run on hardware
//
#include <hip/hip_runtime.h>
#include <hip/hip_bf16.h>

// ---------------------------------------------------------------------------
// MI455X (gfx1250) transformer decoder.
// All GEMMs run on V_WMMA_F32_16X16X32_BF16 (bf16 inputs, fp32 accumulate).
// Interior GEMM blocks use GLOBAL_LOAD_ASYNC_TO_LDS_B128 double-buffering
// (ASYNCcnt) when the toolchain exposes the builtin; guarded fallback keeps
// the sync global->VGPR->ds_store path.
// ---------------------------------------------------------------------------

typedef __attribute__((ext_vector_type(16))) __bf16        v16bf;
typedef __attribute__((ext_vector_type(8)))  float         v8f;
typedef __attribute__((ext_vector_type(4)))  unsigned int  v4u;
typedef int v4i_gcc __attribute__((vector_size(16)));      // builtin param type

#define BB  4
#define SS  512
#define DD  512
#define HH  8
#define DH  64
#define FFF 2048
#define LL  6
#define VV  32000

#define GLOBAL_AS __attribute__((address_space(1)))
#define LDS_AS    __attribute__((address_space(3)))

#if defined(__AMDGCN__) && __has_builtin(__builtin_amdgcn_global_load_async_to_lds_b128)
#define HAS_ASYNC_LDS 1
#else
#define HAS_ASYNC_LDS 0
#endif

__device__ __forceinline__ unsigned short f2bf(float f) {
  unsigned u = __float_as_uint(f);
  u = (u + 0x7FFFu + ((u >> 16) & 1u)) >> 16;   // round-to-nearest-even
  return (unsigned short)u;
}

__device__ __forceinline__ void wait_async0() {
#if defined(__AMDGCN__)
#if __has_builtin(__builtin_amdgcn_s_wait_asynccnt)
  __builtin_amdgcn_s_wait_asynccnt(0);
#else
  asm volatile("s_wait_asynccnt 0x0" ::: "memory");
#endif
#endif
}

// ------------------------------ WMMA GEMM ----------------------------------
// C[M,N] = A[M,K] * B[N,K]^T + bias, optional ReLU, fp32 and/or bf16 output.
// Batched via grid.z:  off = (z/zdiv)*s?1 + (z%zdiv)*s?2  (element offsets).
#define BM 128
#define BN 128
#define BK 32
#define KPAD 40   // row stride (elems): 80 bytes, multiple of 16B for b128 ds ops

union Frag { v4u u[2]; v16bf v; };

// One K-step of WMMA from staged LDS tiles (fragment layouts per ISA 7.12.2).
__device__ __forceinline__ void mma_tile(
    const unsigned short (*As)[KPAD], const unsigned short (*Bs)[KPAD],
    v8f (&acc)[2][4], int wm, int wn, int l16, int hi)
{
  Frag fb[4];
#pragma unroll
  for (int ni = 0; ni < 4; ++ni) {
    const unsigned short* bp = &Bs[wn * 64 + ni * 16 + l16][hi * 8];
    fb[ni].u[0] = *reinterpret_cast<const v4u*>(bp);
    fb[ni].u[1] = *reinterpret_cast<const v4u*>(bp + 16);
  }
#pragma unroll
  for (int mi = 0; mi < 2; ++mi) {
    const unsigned short* ap = &As[wm * 32 + mi * 16 + l16][hi * 8];
    Frag fa;
    fa.u[0] = *reinterpret_cast<const v4u*>(ap);
    fa.u[1] = *reinterpret_cast<const v4u*>(ap + 16);
#pragma unroll
    for (int ni = 0; ni < 4; ++ni) {
      acc[mi][ni] = __builtin_amdgcn_wmma_f32_16x16x32_bf16(
          false, fa.v, false, fb[ni].v, (short)0, acc[mi][ni], false, false);
    }
  }
}

__global__ __launch_bounds__(256) void wmma_gemm_bf16_kernel(
    const unsigned short* __restrict__ A, const unsigned short* __restrict__ B,
    const float* __restrict__ bias, float* __restrict__ C,
    unsigned short* __restrict__ Cb,
    int M, int N, int K, int lda, int ldb, int ldc,
    long long sA1, long long sA2, long long sB1, long long sB2,
    long long sC1, long long sC2, int zdiv, int relu)
{
  __shared__ unsigned short As[2][BM][KPAD];
  __shared__ unsigned short Bs[2][BM][KPAD];

  const int z = blockIdx.z;
  const long long zq = z / zdiv, zr = z % zdiv;
  A += zq * sA1 + zr * sA2;
  B += zq * sB1 + zr * sB2;
  const long long offC = zq * sC1 + zr * sC2;

  const int tid  = threadIdx.x;
  const int wave = tid >> 5, lane = tid & 31;
  const int l16  = lane & 15, hi = lane >> 4;
  const int wm   = wave & 3,  wn = wave >> 2;      // 4 waves on M, 2 on N
  const int row0 = blockIdx.y * BM;
  const int col0 = blockIdx.x * BN;

  v8f acc[2][4] = {};

  const int ldRow = tid >> 2;          // 0..63
  const int ldCol = (tid & 3) << 3;    // 0,8,16,24

#if HAS_ASYNC_LDS
  const bool interior = (row0 + BM <= M) && (col0 + BN <= N) && ((K & (BK - 1)) == 0);
  if (interior) {
    // ---- async double-buffered pipeline (memory -> LDS, no VGPR staging) ----
    auto issue_tile = [&](int k0, int buf) {
#pragma unroll
      for (int rr = 0; rr < 2; ++rr) {
        int r = ldRow + rr * 64;
        __builtin_amdgcn_global_load_async_to_lds_b128(
            (GLOBAL_AS v4i_gcc*)(GLOBAL_AS void*)(A + (long long)(row0 + r) * lda + k0 + ldCol),
            (LDS_AS v4i_gcc*)(LDS_AS void*)(&As[buf][r][ldCol]), 0, 0);
        __builtin_amdgcn_global_load_async_to_lds_b128(
            (GLOBAL_AS v4i_gcc*)(GLOBAL_AS void*)(B + (long long)(col0 + r) * ldb + k0 + ldCol),
            (LDS_AS v4i_gcc*)(LDS_AS void*)(&Bs[buf][r][ldCol]), 0, 0);
      }
    };
    issue_tile(0, 0);
    wait_async0();
    __syncthreads();
    const int nk = K / BK;
    for (int t = 0; t < nk; ++t) {
      const int cur = t & 1;
      if (t + 1 < nk) issue_tile((t + 1) * BK, cur ^ 1);   // prefetch next tile
      mma_tile(As[cur], Bs[cur], acc, wm, wn, l16, hi);    // overlap with async DMA
      wait_async0();                                        // next tile landed
      __syncthreads();                                      // all waves done with cur
    }
  } else
#endif
  {
    // ---- guarded sync path (edge tiles / no async builtin) ----
    for (int k0 = 0; k0 < K; k0 += BK) {
#pragma unroll
      for (int rr = 0; rr < 2; ++rr) {
        int r  = ldRow + rr * 64;
        int ga = row0 + r;
        v4u av = {0u, 0u, 0u, 0u};
        if (ga < M) av = *reinterpret_cast<const v4u*>(A + (long long)ga * lda + k0 + ldCol);
        *reinterpret_cast<v4u*>(&As[0][r][ldCol]) = av;
        int gb = col0 + r;
        v4u bv = {0u, 0u, 0u, 0u};
        if (gb < N) bv = *reinterpret_cast<const v4u*>(B + (long long)gb * ldb + k0 + ldCol);
        *reinterpret_cast<v4u*>(&Bs[0][r][ldCol]) = bv;
      }
      if (k0 + BK < K) {   // speculative prefetch -> global_prefetch_b8
        if (row0 + ldRow < M)
          __builtin_prefetch(A + (long long)(row0 + ldRow) * lda + k0 + BK + ldCol, 0, 0);
        if (col0 + ldRow < N)
          __builtin_prefetch(B + (long long)(col0 + ldRow) * ldb + k0 + BK + ldCol, 0, 0);
      }
      __syncthreads();
      mma_tile(As[0], Bs[0], acc, wm, wn, l16, hi);
      __syncthreads();
    }
  }

#pragma unroll
  for (int mi = 0; mi < 2; ++mi) {
#pragma unroll
    for (int ni = 0; ni < 4; ++ni) {
      int col = col0 + wn * 64 + ni * 16 + l16;
      if (col >= N) continue;
      float bsv = bias ? bias[col] : 0.0f;
      int rbase = row0 + wm * 32 + mi * 16 + hi * 8;  // VGPR j -> M = j (+8 for hi lanes)
#pragma unroll
      for (int j = 0; j < 8; ++j) {
        int rw = rbase + j;
        if (rw >= M) continue;
        float val = acc[mi][ni][j] + bsv;
        if (relu) val = fmaxf(val, 0.0f);
        long long idx = offC + (long long)rw * ldc + col;
        if (C)  C[idx]  = val;
        if (Cb) Cb[idx] = f2bf(val);
      }
    }
  }
}

// --------------------------- helper kernels --------------------------------
__global__ __launch_bounds__(256) void f32_to_bf16_kernel(
    const float* __restrict__ s, unsigned short* __restrict__ d, long long n)
{
  long long i = (long long)blockIdx.x * 256 + threadIdx.x;
  if (i < n) d[i] = f2bf(s[i]);
}

__global__ __launch_bounds__(256) void embed_kernel(
    const int* __restrict__ tokens, const float* __restrict__ emb,
    float* __restrict__ x, unsigned short* __restrict__ xb)
{
  long long i = (long long)blockIdx.x * 256 + threadIdx.x;  // over B*S*D
  int d = (int)(i & (DD - 1));
  long long bs = i >> 9;                                     // D = 512
  int s = (int)(bs & (SS - 1));
  int tok = tokens[bs];
  float e2  = (float)((d >> 1) << 1) / (float)DD;
  float inv = powf(10000.0f, -e2);
  float em  = (float)s * inv;
  float pe  = ((d & 1) == 0) ? sinf(em) : cosf(em);
  float val = emb[(long long)tok * DD + d] * 22.62741699796952f + pe; // sqrt(512)
  x[i] = val;
  xb[i] = f2bf(val);
}

// v [B,S,D] f32  ->  vt [B,H,dh,S] bf16   (so probs*v GEMM reads B as [N,K])
__global__ __launch_bounds__(256) void transpose_v_kernel(
    const float* __restrict__ v, unsigned short* __restrict__ vt)
{
  long long i = (long long)blockIdx.x * 256 + threadIdx.x;  // over B*H*dh*S
  int s = (int)(i & (SS - 1));
  long long r = i >> 9;
  int dh_ = (int)(r & (DH - 1));
  long long r2 = r >> 6;
  int h = (int)(r2 & (HH - 1));
  int b = (int)(r2 >> 3);
  vt[i] = f2bf(v[((long long)b * SS + s) * DD + h * DH + dh_]);
}

// one wave32 per 512-wide row of scores; mode 0: causal+token-pad, 1: enc-pad
__global__ __launch_bounds__(256) void softmax_kernel(
    const float* __restrict__ sc, unsigned short* __restrict__ pr,
    const int* __restrict__ tokens, const float* __restrict__ pmask,
    int mode, float scale)
{
  int wv = threadIdx.x >> 5, lane = threadIdx.x & 31;
  long long row = (long long)blockIdx.x * 8 + wv;    // rows = B*H*S
  int q = (int)(row & (SS - 1));
  int b = (int)(row >> 12);                          // H*S = 4096
  const float* src = sc + row * SS;
  float v[16];
  float mx = -3.0e38f;
#pragma unroll
  for (int i = 0; i < 16; ++i) {
    int kc = lane + (i << 5);
    float s = src[kc] * scale;
    bool m = (mode == 0) ? ((kc > q) || (tokens[b * SS + kc] == 0))
                         : (pmask[b * SS + kc] > 0.5f);
    if (m) s -= 1.0e6f;
    v[i] = s;
    mx = fmaxf(mx, s);
  }
#pragma unroll
  for (int o = 16; o >= 1; o >>= 1) mx = fmaxf(mx, __shfl_xor(mx, o, 32));
  float sum = 0.0f;
#pragma unroll
  for (int i = 0; i < 16; ++i) { v[i] = __expf(v[i] - mx); sum += v[i]; }
#pragma unroll
  for (int o = 16; o >= 1; o >>= 1) sum += __shfl_xor(sum, o, 32);
  float inv = 1.0f / sum;
  unsigned short* dst = pr + row * SS;
#pragma unroll
  for (int i = 0; i < 16; ++i) dst[lane + (i << 5)] = f2bf(v[i] * inv);
}

// y = LayerNorm(x + r)*g + b ; one 256-thread block per D=512 row
__global__ __launch_bounds__(256) void add_ln_kernel(
    const float* __restrict__ x, const float* __restrict__ r,
    const float* __restrict__ g, const float* __restrict__ bt,
    float* __restrict__ y, unsigned short* __restrict__ yb)
{
  long long base = (long long)blockIdx.x * DD;
  int t = threadIdx.x;
  int wv = t >> 5, lane = t & 31;
  __shared__ float red[8];
  __shared__ float stat[2];
  float v0 = x[base + t] + r[base + t];
  float v1 = x[base + t + 256] + r[base + t + 256];
  float s = v0 + v1;
#pragma unroll
  for (int o = 16; o >= 1; o >>= 1) s += __shfl_xor(s, o, 32);
  if (lane == 0) red[wv] = s;
  __syncthreads();
  if (t == 0) {
    float tt = 0.f;
    for (int i = 0; i < 8; ++i) tt += red[i];
    stat[0] = tt * (1.0f / DD);
  }
  __syncthreads();
  float mu = stat[0];
  float d0 = v0 - mu, d1 = v1 - mu;
  float ss = d0 * d0 + d1 * d1;
#pragma unroll
  for (int o = 16; o >= 1; o >>= 1) ss += __shfl_xor(ss, o, 32);
  if (lane == 0) red[wv] = ss;
  __syncthreads();
  if (t == 0) {
    float tt = 0.f;
    for (int i = 0; i < 8; ++i) tt += red[i];
    stat[1] = rsqrtf(tt * (1.0f / DD) + 1e-6f);
  }
  __syncthreads();
  float inv = stat[1];
  float o0 = d0 * inv * g[t] + bt[t];
  float o1 = d1 * inv * g[t + 256] + bt[t + 256];
  y[base + t] = o0;        yb[base + t] = f2bf(o0);
  y[base + t + 256] = o1;  yb[base + t + 256] = f2bf(o1);
}

// ------------------------------ host side ----------------------------------
extern "C" void kernel_launch(void* const* d_in, const int* in_sizes, int n_in,
                              void* d_out, int out_size, void* d_ws, size_t ws_size,
                              hipStream_t stream)
{
  (void)in_sizes; (void)n_in; (void)out_size; (void)ws_size;

  const int*   tokens  = (const int*)d_in[0];
  const float* en_out  = (const float*)d_in[1];
  const float* en_pad  = (const float*)d_in[2];
  const float* emb     = (const float*)d_in[3];
  const float* sa_w    = (const float*)d_in[4];
  const float* sa_b    = (const float*)d_in[5];
  const float* ca_w    = (const float*)d_in[6];
  const float* ca_b    = (const float*)d_in[7];
  const float* ff_w1   = (const float*)d_in[8];
  const float* ff_b1   = (const float*)d_in[9];
  const float* ff_w2   = (const float*)d_in[10];
  const float* ff_b2   = (const float*)d_in[11];
  const float* ln_g    = (const float*)d_in[12];
  const float* ln_b    = (const float*)d_in[13];
  const float* out_w   = (const float*)d_in[14];
  const float* out_b   = (const float*)d_in[15];

  char* wptr = (char*)d_ws;
  auto alloc = [&](size_t bytes) -> char* {
    char* p = wptr;
    wptr += (bytes + 255) & ~(size_t)255;
    return p;
  };

  // bf16 weight copies (fit in 192MB L2 once converted)
  unsigned short* sa_wb  = (unsigned short*)alloc((size_t)LL * 4 * DD * DD * 2);
  unsigned short* ca_wb  = (unsigned short*)alloc((size_t)LL * 4 * DD * DD * 2);
  unsigned short* f1_wb  = (unsigned short*)alloc((size_t)LL * FFF * DD * 2);
  unsigned short* f2_wb  = (unsigned short*)alloc((size_t)LL * DD * FFF * 2);
  unsigned short* out_wb = (unsigned short*)alloc((size_t)VV * DD * 2);
  unsigned short* en_bf  = (unsigned short*)alloc((size_t)BB * SS * DD * 2);
  // activations
  float*          xf    = (float*)alloc((size_t)BB * SS * DD * 4);
  unsigned short* xb    = (unsigned short*)alloc((size_t)BB * SS * DD * 2);
  float*          hf    = (float*)alloc((size_t)BB * SS * DD * 4);
  unsigned short* hb    = (unsigned short*)alloc((size_t)BB * SS * DD * 2);
  float*          af    = (float*)alloc((size_t)BB * SS * DD * 4);
  unsigned short* ab    = (unsigned short*)alloc((size_t)BB * SS * DD * 2);
  unsigned short* qb    = (unsigned short*)alloc((size_t)BB * SS * DD * 2);
  unsigned short* kb    = (unsigned short*)alloc((size_t)BB * SS * DD * 2);
  float*          vf    = (float*)alloc((size_t)BB * SS * DD * 4);
  unsigned short* vt    = (unsigned short*)alloc((size_t)BB * HH * DH * SS * 2);
  unsigned short* attnb = (unsigned short*)alloc((size_t)BB * SS * DD * 2);
  float*          projf = (float*)alloc((size_t)BB * SS * DD * 4);
  unsigned short* f1b   = (unsigned short*)alloc((size_t)BB * SS * FFF * 2);
  // big transients live inside d_out (262MB; final GEMM overwrites it last)
  float*          scores = (float*)d_out;
  unsigned short* probs  = (unsigned short*)((char*)d_out + (size_t)BB * HH * SS * SS * 4);

  auto conv = [&](const float* s, unsigned short* d, long long n) {
    f32_to_bf16_kernel<<<(unsigned)((n + 255) / 256), 256, 0, stream>>>(s, d, n);
  };
  auto gemm = [&](const unsigned short* A, const unsigned short* Bm, const float* bias,
                  float* C, unsigned short* Cb, int M, int N, int K,
                  int lda, int ldb, int ldc,
                  long long a1, long long a2, long long b1, long long b2,
                  long long c1, long long c2, int zdiv, int nz, int relu) {
    dim3 g((N + BN - 1) / BN, (M + BM - 1) / BM, nz);
    wmma_gemm_bf16_kernel<<<g, 256, 0, stream>>>(A, Bm, bias, C, Cb, M, N, K,
                                                 lda, ldb, ldc, a1, a2, b1, b2,
                                                 c1, c2, zdiv, relu);
  };

  // ---- one-time conversions + embedding ----
  conv(sa_w,  sa_wb,  (long long)LL * 4 * DD * DD);
  conv(ca_w,  ca_wb,  (long long)LL * 4 * DD * DD);
  conv(ff_w1, f1_wb,  (long long)LL * FFF * DD);
  conv(ff_w2, f2_wb,  (long long)LL * DD * FFF);
  conv(out_w, out_wb, (long long)VV * DD);
  conv(en_out, en_bf, (long long)BB * SS * DD);
  embed_kernel<<<(BB * SS * DD) / 256, 256, 0, stream>>>(tokens, emb, xf, xb);

  const int MBS = BB * SS;  // 2048 token rows
  const long long sSD = (long long)SS * DD;
  const long long sHSS = (long long)HH * SS * SS;
  const long long sSSq = (long long)SS * SS;
  const long long sHDS = (long long)HH * DH * SS;
  const long long sDS  = (long long)DH * SS;

  for (int i = 0; i < LL; ++i) {
    const unsigned short* sw = sa_wb + (size_t)i * 4 * DD * DD;
    const float* sb = sa_b + (size_t)i * 4 * DD;
    // ---- self attention ----
    gemm(xb, sw + 0 * DD * DD, sb + 0 * DD, nullptr, qb, MBS, DD, DD, DD, DD, DD, 0,0,0,0,0,0, 1, 1, 0);
    gemm(xb, sw + 1 * DD * DD, sb + 1 * DD, nullptr, kb, MBS, DD, DD, DD, DD, DD, 0,0,0,0,0,0, 1, 1, 0);
    gemm(xb, sw + 2 * DD * DD, sb + 2 * DD, vf, nullptr, MBS, DD, DD, DD, DD, DD, 0,0,0,0,0,0, 1, 1, 0);
    transpose_v_kernel<<<(BB * HH * DH * SS) / 256, 256, 0, stream>>>(vf, vt);
    gemm(qb, kb, nullptr, scores, nullptr, SS, SS, DH, DD, DD, SS,
         sSD, DH, sSD, DH, sHSS, sSSq, HH, BB * HH, 0);
    softmax_kernel<<<(BB * HH * SS) / 8, 256, 0, stream>>>(scores, probs, tokens, en_pad, 0, 0.125f);
    gemm(probs, vt, nullptr, nullptr, attnb, SS, DH, SS, SS, SS, DD,
         sHSS, sSSq, sHDS, sDS, sSD, DH, HH, BB * HH, 0);
    gemm(attnb, sw + 3 * DD * DD, sb + 3 * DD, projf, nullptr, MBS, DD, DD, DD, DD, DD, 0,0,0,0,0,0, 1, 1, 0);
    add_ln_kernel<<<MBS, 256, 0, stream>>>(xf, projf,
        ln_g + ((size_t)i * 3 + 0) * DD, ln_b + ((size_t)i * 3 + 0) * DD, hf, hb);
    // ---- cross attention ----
    const unsigned short* cw = ca_wb + (size_t)i * 4 * DD * DD;
    const float* cb = ca_b + (size_t)i * 4 * DD;
    gemm(hb,    cw + 0 * DD * DD, cb + 0 * DD, nullptr, qb, MBS, DD, DD, DD, DD, DD, 0,0,0,0,0,0, 1, 1, 0);
    gemm(en_bf, cw + 1 * DD * DD, cb + 1 * DD, nullptr, kb, MBS, DD, DD, DD, DD, DD, 0,0,0,0,0,0, 1, 1, 0);
    gemm(en_bf, cw + 2 * DD * DD, cb + 2 * DD, vf, nullptr, MBS, DD, DD, DD, DD, DD, 0,0,0,0,0,0, 1, 1, 0);
    transpose_v_kernel<<<(BB * HH * DH * SS) / 256, 256, 0, stream>>>(vf, vt);
    gemm(qb, kb, nullptr, scores, nullptr, SS, SS, DH, DD, DD, SS,
         sSD, DH, sSD, DH, sHSS, sSSq, HH, BB * HH, 0);
    softmax_kernel<<<(BB * HH * SS) / 8, 256, 0, stream>>>(scores, probs, tokens, en_pad, 1, 0.125f);
    gemm(probs, vt, nullptr, nullptr, attnb, SS, DH, SS, SS, SS, DD,
         sHSS, sSSq, sHDS, sDS, sSD, DH, HH, BB * HH, 0);
    gemm(attnb, cw + 3 * DD * DD, cb + 3 * DD, projf, nullptr, MBS, DD, DD, DD, DD, DD, 0,0,0,0,0,0, 1, 1, 0);
    add_ln_kernel<<<MBS, 256, 0, stream>>>(hf, projf,
        ln_g + ((size_t)i * 3 + 1) * DD, ln_b + ((size_t)i * 3 + 1) * DD, af, ab);
    // ---- feed forward ----
    gemm(ab, f1_wb + (size_t)i * FFF * DD, ff_b1 + (size_t)i * FFF, nullptr, f1b,
         MBS, FFF, DD, DD, DD, FFF, 0,0,0,0,0,0, 1, 1, 1 /*relu*/);
    gemm(f1b, f2_wb + (size_t)i * DD * FFF, ff_b2 + (size_t)i * DD, projf, nullptr,
         MBS, DD, FFF, FFF, FFF, DD, 0,0,0,0,0,0, 1, 1, 0);
    add_ln_kernel<<<MBS, 256, 0, stream>>>(af, projf,
        ln_g + ((size_t)i * 3 + 2) * DD, ln_b + ((size_t)i * 3 + 2) * DD, xf, xb);
  }

  // ---- vocab projection (dominant GEMM: 2048 x 32000 x 512) ----
  gemm(xb, out_wb, out_b, (float*)d_out, nullptr, MBS, VV, DD, DD, DD, VV,
       0,0,0,0,0,0, 1, 1, 0);
}
